// D3QN_62526133895332
// MI455X (gfx1250) — compile-verified, matrix operand order checked by reading, not run on hardware
//
#include <hip/hip_runtime.h>
#include <hip/hip_bf16.h>
#include <math.h>

typedef __attribute__((ext_vector_type(16))) _Float16 v16h;
typedef __attribute__((ext_vector_type(8)))  _Float16 v8h;
typedef __attribute__((ext_vector_type(8)))  float    v8f;

struct NetParams { const float *b1, *b2, *l1b, *l2b, *l3b; };
struct WSrcs     { const float *W1, *W2, *L1w, *L2w, *L3w; };

#define NBATCH   4096
#define XG_N     (4096LL*64*32)      // f16 elems: [b][pixel][ci(32, 19 real)]
#define WT_PER_NET 95232             // f16 elems per net
// per-net weight region offsets (f16 elems)
#define C1_OFF 0        // [32][800]  k = tap*32 + ci
#define C2_OFF 25600    // [64][288]  k = tap*32 + ci
#define L1_OFF 44032    // [128][256]
#define L2_OFF 76800    // [112][128] rows >=100 zero
#define L3_OFF 91136    // [32][128]  rows >= out_dim zero, k>=100 zero

static __device__ __forceinline__ v16h cat16(v8h lo, v8h hi) {
  return __builtin_shufflevector(lo, hi, 0,1,2,3,4,5,6,7,8,9,10,11,12,13,14,15);
}
static __device__ __forceinline__ v8f wmma16(v16h a, v16h b, v8f c) {
  return __builtin_amdgcn_wmma_f32_16x16x32_f16(false, a, false, b, (short)0, c, false, false);
}
static __device__ __forceinline__ v16h ldA(const _Float16* row, int half) {
  v8h lo = *(const v8h*)(row + 8*half);
  v8h hi = *(const v8h*)(row + 16 + 8*half);
  return cat16(lo, hi);
}

// ---------------------------------------------------------------------------
// Kernel 1: prep — f16 channel-last board copy + weight transpose/pad
//           (also zero-fills the 32-elem OOB zero row at xg[XG_N..XG_N+32))
// ---------------------------------------------------------------------------
__global__ __launch_bounds__(256) void prep_kernel(
    const float* __restrict__ xb, _Float16* __restrict__ xg,
    _Float16* __restrict__ wt, WSrcs s0, WSrcs s1)
{
  long long total = XG_N + 2LL*WT_PER_NET + 32;
  for (long long i = (long long)blockIdx.x*256 + threadIdx.x; i < total;
       i += (long long)gridDim.x*256) {
    if (i < XG_N) {
      int c = (int)(i & 31); long long t = i >> 5;
      int p = (int)(t & 63); long long b = t >> 6;
      float v = (c < 19) ? xb[b*1216 + c*64 + p] : 0.f;
      xg[i] = (_Float16)v;
    } else if (i < XG_N + 2LL*WT_PER_NET) {
      long long w = i - XG_N;
      int net = (int)(w / WT_PER_NET);
      int r   = (int)(w % WT_PER_NET);
      WSrcs s = net ? s1 : s0;
      float v = 0.f;
      if (r < C2_OFF) {                       // conv1 weights
        int row = r / 800, k = r % 800, tap = k >> 5, ci = k & 31;
        if (ci < 19) v = s.W1[(row*19 + ci)*25 + tap];
      } else if (r < L1_OFF) {                // conv2 weights
        int r2 = r - C2_OFF;
        int row = r2 / 288, k = r2 % 288, tap = k >> 5, ci = k & 31;
        v = s.W2[(row*32 + ci)*9 + tap];
      } else if (r < L2_OFF) {                // L1: [256,128] -> [n][k]
        int r3 = r - L1_OFF; int n = r3 >> 8, k = r3 & 255;
        v = s.L1w[k*128 + n];
      } else if (r < L3_OFF) {                // L2: [128,100] -> [n][k], n<112
        int r4 = r - L2_OFF; int n = r4 >> 7, k = r4 & 127;
        if (n < 100) v = s.L2w[k*100 + n];
      } else {                                // L3: [100,od] -> [n][k]
        int r5 = r - L3_OFF; int n = r5 >> 7, k = r5 & 127;
        int od = net ? 19 : 1;
        if (n < od && k < 100) v = s.L3w[k*od + n];
      }
      wt[w] = (_Float16)v;
    } else {
      xg[i - 2LL*WT_PER_NET] = (_Float16)0.f;   // zero row at xg + XG_N
    }
  }
}

// ---------------------------------------------------------------------------
// Kernel 2: fused CNN (state + piece nets), 16 samples per workgroup.
// Branchless conv taps: OOB taps read a zero row (address cndmask), so every
// WMMA executes with full EXEC and accumulators stay in-place.
// ---------------------------------------------------------------------------
__global__ __launch_bounds__(256) void fused_cnn_kernel(
    const _Float16* __restrict__ xg, const _Float16* __restrict__ wt,
    NetParams np0, NetParams np1,
    float* __restrict__ qstate, float* __restrict__ plog)
{
  __shared__ _Float16 pool1[16*16*32];   // [s][4][4][32]   16 KB
  __shared__ _Float16 feat [16*256];     // [s][256]         8 KB
  __shared__ _Float16 a1   [16*128];     // [s][128]         4 KB
  __shared__ _Float16 a2   [16*128];     // [s][128]         4 KB
  __shared__ _Float16 zrow [32];         // LDS zero row for conv2 OOB taps

  const int tid  = threadIdx.x;
  const int wave = tid >> 5;
  const int lane = tid & 31;
  const int half = lane >> 4;
  const int nl   = lane & 15;
  const int b0   = blockIdx.x * 16;
  const _Float16* zg = xg + XG_N;        // global zero row for conv1 OOB taps

  if (tid < 32) zrow[tid] = (_Float16)0.f;

  for (int net = 0; net < 2; ++net) {
    const _Float16* W   = wt + (long long)net * WT_PER_NET;
    const _Float16* c1w = W + C1_OFF;
    const _Float16* c2w = W + C2_OFF;
    const _Float16* l1w = W + L1_OFF;
    const _Float16* l2w = W + L2_OFF;
    const _Float16* l3w = W + L3_OFF;
    NetParams np = net ? np1 : np0;

    __syncthreads();   // protect LDS reuse across net iterations (+ zrow init)

    // ---- conv1 (19->32, 5x5 pad2) + maxpool2 -> pool1[s][4][4][32] ----
    for (int c = wave; c < 32; c += 8) {
      int ntile = c >> 4, pp = c & 15;
      int ppy = pp >> 2, ppx = pp & 3;
      int ng = ntile*16 + nl;
      v8f acc0 = {}, acc1 = {}, acc2 = {}, acc3 = {};
      const _Float16* xrow0 = xg + (long long)(b0 + nl)*64*32;
      for (int tap = 0; tap < 25; ++tap) {
        int ty = tap / 5, tx = tap % 5;
        v16h bf = ldA(c1w + ng*800 + tap*32, half);
        #pragma unroll
        for (int sp = 0; sp < 4; ++sp) {
          int py = 2*ppy + (sp >> 1), px = 2*ppx + (sp & 1);
          int qy = py + ty - 2, qx = px + tx - 2;
          bool ok = ((unsigned)qy < 8u) & ((unsigned)qx < 8u);
          const _Float16* ap = ok ? (xrow0 + (qy*8 + qx)*32) : zg;
          v16h af = ldA(ap, half);
          if      (sp == 0) acc0 = wmma16(af, bf, acc0);
          else if (sp == 1) acc1 = wmma16(af, bf, acc1);
          else if (sp == 2) acc2 = wmma16(af, bf, acc2);
          else              acc3 = wmma16(af, bf, acc3);
        }
      }
      float bias = np.b1[ng];
      #pragma unroll
      for (int v = 0; v < 8; ++v) {
        float mx = fmaxf(fmaxf(acc0[v], acc1[v]), fmaxf(acc2[v], acc3[v]));
        float val = fmaxf(mx + bias, 0.f);
        int m = v + 8*half;
        pool1[((m*4 + ppy)*4 + ppx)*32 + ng] = (_Float16)val;
      }
    }
    __syncthreads();

    // ---- conv2 (32->64, 3x3 pad1) + maxpool2 -> feat[s][c*4+y*2+x] ----
    for (int c = wave; c < 16; c += 8) {
      int ntile = c >> 2, pp = c & 3;
      int ppy = pp >> 1, ppx = pp & 1;
      int ng = ntile*16 + nl;
      v8f acc0 = {}, acc1 = {}, acc2 = {}, acc3 = {};
      for (int tap = 0; tap < 9; ++tap) {
        int ty = tap / 3, tx = tap % 3;
        v16h bf = ldA(c2w + ng*288 + tap*32, half);
        #pragma unroll
        for (int sp = 0; sp < 4; ++sp) {
          int py = 2*ppy + (sp >> 1), px = 2*ppx + (sp & 1);
          int qy = py + ty - 1, qx = px + tx - 1;
          bool ok = ((unsigned)qy < 4u) & ((unsigned)qx < 4u);
          const _Float16* ap = ok ? &pool1[((nl*4 + qy)*4 + qx)*32] : zrow;
          v16h af = ldA(ap, half);
          if      (sp == 0) acc0 = wmma16(af, bf, acc0);
          else if (sp == 1) acc1 = wmma16(af, bf, acc1);
          else if (sp == 2) acc2 = wmma16(af, bf, acc2);
          else              acc3 = wmma16(af, bf, acc3);
        }
      }
      float bias = np.b2[ng];
      #pragma unroll
      for (int v = 0; v < 8; ++v) {
        float mx = fmaxf(fmaxf(acc0[v], acc1[v]), fmaxf(acc2[v], acc3[v]));
        float val = fmaxf(mx + bias, 0.f);
        int m = v + 8*half;
        feat[m*256 + ng*4 + ppy*2 + ppx] = (_Float16)val;  // (C,H,W) flatten
      }
    }
    __syncthreads();

    // ---- L1: [16,256] @ [256,128], relu -> a1 ----
    {
      int ng = wave*16 + nl;           // 8 waves cover 128 outputs
      v8f acc = {};
      #pragma unroll
      for (int kc = 0; kc < 8; ++kc) {
        v16h af = ldA(&feat[nl*256 + kc*32], half);
        v16h bf = ldA(l1w + ng*256 + kc*32, half);
        acc = wmma16(af, bf, acc);
      }
      float bias = np.l1b[ng];
      #pragma unroll
      for (int v = 0; v < 8; ++v) {
        int m = v + 8*half;
        a1[m*128 + ng] = (_Float16)fmaxf(acc[v] + bias, 0.f);
      }
    }
    for (int i = tid; i < 16*128; i += 256) a2[i] = (_Float16)0.f;
    __syncthreads();

    // ---- L2: [16,128] @ [128,100], relu -> a2 (padded to 128) ----
    if (wave < 7) {
      int ng = wave*16 + nl;           // 112 padded outputs, 100 real
      v8f acc = {};
      #pragma unroll
      for (int kc = 0; kc < 4; ++kc) {
        v16h af = ldA(&a1[nl*128 + kc*32], half);
        v16h bf = ldA(l2w + ng*128 + kc*32, half);
        acc = wmma16(af, bf, acc);
      }
      float bias = (ng < 100) ? np.l2b[ng] : 0.f;
      #pragma unroll
      for (int v = 0; v < 8; ++v) {
        int m = v + 8*half;
        a2[m*128 + ng] = (_Float16)fmaxf(acc[v] + bias, 0.f);
      }
    }
    __syncthreads();

    // ---- L3 ----
    if (net == 0) {
      if (wave == 0) {                 // out_dim = 1 -> sigmoid -> q_state
        v8f acc = {};
        #pragma unroll
        for (int kc = 0; kc < 4; ++kc) {
          v16h af = ldA(&a2[nl*128 + kc*32], half);
          v16h bf = ldA(l3w + nl*128 + kc*32, half);
          acc = wmma16(af, bf, acc);
        }
        if (nl == 0) {
          float bias = np.l3b[0];
          #pragma unroll
          for (int v = 0; v < 8; ++v) {
            int m = v + 8*half;
            float z = acc[v] + bias;
            qstate[b0 + m] = 1.f / (1.f + expf(-z));
          }
        }
      }
    } else {
      if (wave < 2) {                  // out_dim = 19 -> piece logits
        int ng = wave*16 + nl;
        v8f acc = {};
        #pragma unroll
        for (int kc = 0; kc < 4; ++kc) {
          v16h af = ldA(&a2[nl*128 + kc*32], half);
          v16h bf = ldA(l3w + ng*128 + kc*32, half);
          acc = wmma16(af, bf, acc);
        }
        if (ng < 19) {
          float bias = np.l3b[ng];
          #pragma unroll
          for (int v = 0; v < 8; ++v) {
            int m = v + 8*half;
            plog[(b0 + m)*19 + ng] = acc[v] + bias;
          }
        }
      }
    }
  }
}

// ---------------------------------------------------------------------------
// Kernel 3: per-column (over batch!) softmax stats
// ---------------------------------------------------------------------------
__global__ __launch_bounds__(256) void colstats_kernel(
    const float* __restrict__ pl, float* __restrict__ cm, float* __restrict__ cs)
{
  int e = blockIdx.x, t = threadIdx.x;
  __shared__ float red[256];
  float mx = -INFINITY;
  for (int b = t; b < NBATCH; b += 256) mx = fmaxf(mx, pl[b*19 + e]);
  red[t] = mx; __syncthreads();
  for (int s = 128; s > 0; s >>= 1) {
    if (t < s) red[t] = fmaxf(red[t], red[t + s]);
    __syncthreads();
  }
  float M = red[0]; __syncthreads();
  float sm = 0.f;
  for (int b = t; b < NBATCH; b += 256) sm += expf(pl[b*19 + e] - M);
  red[t] = sm; __syncthreads();
  for (int s = 128; s > 0; s >>= 1) {
    if (t < s) red[t] += red[t + s];
    __syncthreads();
  }
  if (t == 0) { cm[e] = M; cs[e] = red[0]; }
}

// ---------------------------------------------------------------------------
// Kernel 4: finalize -> q_pieces, p_id (argmax, first-max), q_board
// ---------------------------------------------------------------------------
__global__ __launch_bounds__(128) void finalize_kernel(
    const float* __restrict__ xp, const float* __restrict__ mask,
    const float* __restrict__ pl, const float* __restrict__ qs,
    const float* __restrict__ cm, const float* __restrict__ cs,
    float* __restrict__ out)
{
  int b = blockIdx.x, t = threadIdx.x;
  __shared__ float availf[19];
  __shared__ float qv[19];
  __shared__ int pid;
  if (t < 19) availf[t] = 0.f;
  __syncthreads();
  const float* mrow = mask + (size_t)b * 1900;
  for (int i = t; i < 1900; i += 128)
    if (mrow[i] != 0.f) availf[i / 100] = 1.f;   // benign same-value race
  __syncthreads();
  float q_s = qs[b];
  if (t < 19) {
    float l = pl[b*19 + t];
    float q = expf(l - cm[t]) / cs[t] * xp[b*19 + t] * availf[t];
    float qp = 0.5f*q_s + 0.5f*q;
    qv[t] = qp;
    out[b*19 + t] = qp;
  }
  __syncthreads();
  if (t == 0) {
    int best = 0; float bv = qv[0];
    for (int e = 1; e < 19; ++e) if (qv[e] > bv) { bv = qv[e]; best = e; }
    pid = best;
  }
  __syncthreads();
  // softmax over singleton axis == 1.0 -> q_board = 0.5*q_state + 0.5*mask
  if (t < 100)
    out[NBATCH*19 + b*100 + t] = 0.5f*q_s + 0.5f*mrow[pid*100 + t];
}

// ---------------------------------------------------------------------------
extern "C" void kernel_launch(void* const* d_in, const int* in_sizes, int n_in,
                              void* d_out, int out_size, void* d_ws, size_t ws_size,
                              hipStream_t stream) {
  (void)in_sizes; (void)n_in; (void)out_size; (void)ws_size;
  const float* xp = (const float*)d_in[0];
  const float* xb = (const float*)d_in[1];
  const float* xm = (const float*)d_in[2];
  // state params: 3..12  (W1,b1,W2,b2,L1w,L1b,L2w,L2b,L3w,L3b); piece: 13..22
  WSrcs s0 { (const float*)d_in[3],  (const float*)d_in[5],  (const float*)d_in[7],
             (const float*)d_in[9],  (const float*)d_in[11] };
  WSrcs s1 { (const float*)d_in[13], (const float*)d_in[15], (const float*)d_in[17],
             (const float*)d_in[19], (const float*)d_in[21] };
  NetParams np0 { (const float*)d_in[4],  (const float*)d_in[6],  (const float*)d_in[8],
                  (const float*)d_in[10], (const float*)d_in[12] };
  NetParams np1 { (const float*)d_in[14], (const float*)d_in[16], (const float*)d_in[18],
                  (const float*)d_in[20], (const float*)d_in[22] };
  // pos_params (d_in[23..32]) are provably dead: softmax over a singleton axis
  // is identically 1, so the expert outputs never reach q_board.

  char* ws = (char*)d_ws;
  float*     pl  = (float*)(ws + 0);          // 4096*19 f32
  float*     qsb = (float*)(ws + 311296);     // 4096 f32
  float*     cm  = (float*)(ws + 327680);     // 19 f32 (pad 128)
  float*     cs  = (float*)(ws + 327808);     // 19 f32 (pad 128)
  _Float16*  wt  = (_Float16*)(ws + 327936);  // 2*95232 f16
  _Float16*  xg  = (_Float16*)(ws + 712704);  // 4096*64*32 f16 + 32 zero row

  prep_kernel<<<4096, 256, 0, stream>>>(xb, xg, wt, s0, s1);
  fused_cnn_kernel<<<NBATCH/16, 256, 0, stream>>>(xg, wt, np0, np1, qsb, pl);
  colstats_kernel<<<19, 256, 0, stream>>>(pl, cm, cs);
  finalize_kernel<<<NBATCH, 128, 0, stream>>>(xp, xm, pl, qsb, cm, cs, (float*)d_out);
}
